// NodeEmbedGNN_3685081940614
// MI455X (gfx1250) — matrix operand order repaired and checked by reading, not compiled, not run on hardware
//
#include <hip/hip_runtime.h>

// ---------------------------------------------------------------------------
// NodeEmbedGNN for MI455X (gfx1250).
// Memory-bound (edge state = 205 MB per E x 32 tensor); GEMMs run on the WMMA
// pipe (v_wmma_f32_16x16x32_bf16, f32 accumulate, K=32 = full inner dim in one
// instruction). The edge path fuses GEMM + gated message scatter + src/dst
// gathers + BN-stat reduction into a single pass over w, eliminating one full
// read of w and the w1 write->read round trip per layer.
// ---------------------------------------------------------------------------

#define NN      100000      // nodes
#define NE      1600000     // edges
#define DEPTH   12
#define UNITS   32
#define BN_EPS  1e-5f

typedef __attribute__((ext_vector_type(8)))  float  v8f;
typedef __attribute__((ext_vector_type(16))) __bf16 v16bf;

__device__ __forceinline__ float sigmoid_f(float x) { return 1.0f / (1.0f + __expf(-x)); }
__device__ __forceinline__ float silu_f(float x)    { return x * sigmoid_f(x); }

// ---------------------------------------------------------------- utilities
__global__ void zero_kernel(float* __restrict__ p, int n) {
    int i = blockIdx.x * blockDim.x + threadIdx.x;
    if (i < n) p[i] = 0.0f;
}

// degree count: cnt[src[e]] += 1
__global__ void deg_count_kernel(const int* __restrict__ ei, float* __restrict__ cnt) {
    int e = blockIdx.x * blockDim.x + threadIdx.x;
    if (e < NE) atomicAdd(&cnt[ei[e]], 1.0f);
}

// cnt -> 1/max(cnt,1) in place
__global__ void inv_deg_kernel(float* __restrict__ cnt) {
    int n = blockIdx.x * blockDim.x + threadIdx.x;
    if (n < NN) cnt[n] = 1.0f / fmaxf(cnt[n], 1.0f);
}

// ------------------------------------------------------ input projections
// h = silu(x @ v_w0 + v_b0)   [N,3]x[3,32]
__global__ void init_h_kernel(const float* __restrict__ x, const float* __restrict__ w0,
                              const float* __restrict__ b0, float* __restrict__ h) {
    int idx = blockIdx.x * blockDim.x + threadIdx.x;   // over N*32
    int n = idx >> 5, u = idx & 31;
    float acc = b0[u];
#pragma unroll
    for (int f = 0; f < 3; ++f) acc += x[n * 3 + f] * w0[f * UNITS + u];
    h[idx] = silu_f(acc);
}

// w = silu(edge_attr @ e_w0 + e_b0)   [E,1]x[1,32]
__global__ void init_w_kernel(const float* __restrict__ ea, const float* __restrict__ w0,
                              const float* __restrict__ b0, float* __restrict__ w) {
    int idx = blockIdx.x * blockDim.x + threadIdx.x;   // over E*32
    int e = idx >> 5, u = idx & 31;
    w[idx] = silu_f(b0[u] + ea[e] * w0[u]);
}

// --------------------------------------------- weight -> B-fragment packing
// B (32x16 half of a 32x32 weight) per-lane layout for v_wmma_*_16x16x32:
//   lane L<16 : b[j] = W[j<8 ? j : j+8][16h+L],     j=0..15
//   lane L>=16: b[j] = W[j<8 ? 8+j : 16+j][16h+L-16]
// task id = (d*5 + m)*2 + half ; element = task*32 + lane
__global__ void pack_weights_kernel(const float* __restrict__ vW1, const float* __restrict__ vW2,
                                    const float* __restrict__ vW3, const float* __restrict__ vW4,
                                    const float* __restrict__ eW,  v16bf* __restrict__ frag) {
    int task = blockIdx.x;              // 0 .. DEPTH*5*2-1
    int half = task & 1;
    int dm   = task >> 1;
    int m    = dm % 5;
    int d    = dm / 5;
    const float* W;
    switch (m) {
        case 0: W = vW1; break; case 1: W = vW2; break;
        case 2: W = vW3; break; case 3: W = vW4; break;
        default: W = eW; break;
    }
    W += (size_t)d * UNITS * UNITS;
    int L = threadIdx.x;                // 0..31
    int nb = half * 16;
    v16bf b;
    if (L < 16) {
#pragma unroll
        for (int j = 0; j < 16; ++j) b[j] = (__bf16)W[((j < 8) ? j : j + 8) * UNITS + nb + L];
    } else {
#pragma unroll
        for (int j = 0; j < 16; ++j) b[j] = (__bf16)W[((j < 8) ? 8 + j : 16 + j) * UNITS + nb + (L - 16)];
    }
    frag[(size_t)task * 32 + L] = b;
}

// pack node biases into [D][4][32] so node_gemm avoids divergent pointer selects
__global__ void pack_bias_kernel(const float* __restrict__ vB1, const float* __restrict__ vB2,
                                 const float* __restrict__ vB3, const float* __restrict__ vB4,
                                 float* __restrict__ bp) {
    int d = blockIdx.x;            // 0..DEPTH-1
    int t = threadIdx.x;           // 0..127
    int m = t >> 5, u = t & 31;
    const float* B = (m == 0) ? vB1 : (m == 1) ? vB2 : (m == 2) ? vB3 : vB4;
    bp[d * 4 * UNITS + t] = B[d * UNITS + u];
}

// --------------------------------------------------------- WMMA GEMM bodies
// A fragment (16x32, bf16, §7.12.2): lane L<16 -> row tile*16+L, K 0..7 & 16..23
//                                    lane>=16  -> row tile*16+L-16, K 8..15 & 24..31
__device__ __forceinline__ v16bf load_a_frag(const float* __restrict__ src, int tile, int lane) {
    int r   = tile * 16 + (lane & 15);
    int klo = (lane < 16) ? 0 : 8;
    const float* ap = src + (size_t)r * UNITS + klo;
    v16bf a;
#pragma unroll
    for (int i = 0; i < 8; ++i) {
        a[i]     = (__bf16)ap[i];
        a[8 + i] = (__bf16)ap[16 + i];
    }
    return a;
}

// x1..x4 = h @ vW[1..4] + vB ; 8 waves/block, wave -> (tile, mat)
__global__ void node_gemm_kernel(const float* __restrict__ h, const v16bf* __restrict__ fragL,
                                 const float* __restrict__ biasL,   // [4][32] for this layer
                                 float* __restrict__ xbuf) {
    int lane = threadIdx.x & 31;
    int wv   = threadIdx.x >> 5;
    int task = blockIdx.x * 8 + wv;          // NN/16 * 4 tasks, exact multiple of 8
    int tile = task >> 2;
    int m    = task & 3;

    v16bf a  = load_a_frag(h, tile, lane);
    const v16bf* fb = fragL + (size_t)m * 64;
    v16bf b0 = fb[lane];
    v16bf b1 = fb[32 + lane];

    v8f c0 = {}, c1 = {};
    c0 = __builtin_amdgcn_wmma_f32_16x16x32_bf16(false, a, false, b0, (short)0, c0, false, false);
    c1 = __builtin_amdgcn_wmma_f32_16x16x32_bf16(false, a, false, b1, (short)0, c1, false, false);

    const float* bias = biasL + m * UNITS;
    int col  = lane & 15;
    int row0 = tile * 16 + ((lane >> 4) << 3);   // +0 for lanes 0-15, +8 for 16-31
    float bb0 = bias[col], bb1 = bias[16 + col];
    float* out = xbuf + (size_t)m * NN * UNITS;
#pragma unroll
    for (int r = 0; r < 8; ++r) {
        out[(size_t)(row0 + r) * UNITS + col]      = c0[r] + bb0;
        out[(size_t)(row0 + r) * UNITS + 16 + col] = c1[r] + bb1;
    }
}

// ---------------------------------------------------------------------------
// Fused edge pass: one read of w per layer.
//   * WMMA: w1 = w @ eW
//   * gated messages: agg[src] += sigmoid(w) * x2[dst]   (from the A-frag regs)
//   * t = w1 + eB + x3[src] + x4[dst]  -> stored to w1, BN stats reduced
// Lanes L and L+16 together hold all 32 units of edge row L of the tile.
// ---------------------------------------------------------------------------
__global__ void edge_fused_kernel(const float* __restrict__ w, const v16bf* __restrict__ fragE,
                                  const float* __restrict__ eB,
                                  const float* __restrict__ x2, const float* __restrict__ x3,
                                  const float* __restrict__ x4, const int* __restrict__ ei,
                                  float* __restrict__ agg, float* __restrict__ w1,
                                  float* __restrict__ stats) {
    __shared__ float s[2 * UNITS];
    if (threadIdx.x < 2 * UNITS) s[threadIdx.x] = 0.0f;
    __syncthreads();

    int lane = threadIdx.x & 31;
    int wv   = threadIdx.x >> 5;
    int tile = blockIdx.x * 8 + wv;          // NE/16 tiles, exact multiple of 8

    // ---- load this lane's half-row of w: keep f32 for gating, bf16 for WMMA
    int r   = tile * 16 + (lane & 15);
    int klo = (lane < 16) ? 0 : 8;
    const float* ap = w + (size_t)r * UNITS + klo;
    float f[16];
    v16bf a;
#pragma unroll
    for (int i = 0; i < 8; ++i) {
        f[i]     = ap[i];
        f[8 + i] = ap[16 + i];
        a[i]     = (__bf16)f[i];
        a[8 + i] = (__bf16)f[8 + i];
    }

    // ---- gated message + scatter for the 16 units this lane holds of edge r
    int s_e = ei[r], d_e = ei[NE + r];
    const float* x2d  = x2 + (size_t)d_e * UNITS;
    float*       aggs = agg + (size_t)s_e * UNITS;
#pragma unroll
    for (int i = 0; i < 8; ++i) {
        int u0 = klo + i, u1 = klo + 16 + i;
        atomicAdd(&aggs[u0], sigmoid_f(f[i])     * x2d[u0]);
        atomicAdd(&aggs[u1], sigmoid_f(f[8 + i]) * x2d[u1]);
    }

    // ---- WMMA: w1 = w @ eW
    v16bf b0 = fragE[lane];
    v16bf b1 = fragE[32 + lane];
    v8f c0 = {}, c1 = {};
    c0 = __builtin_amdgcn_wmma_f32_16x16x32_bf16(false, a, false, b0, (short)0, c0, false, false);
    c1 = __builtin_amdgcn_wmma_f32_16x16x32_bf16(false, a, false, b1, (short)0, c1, false, false);

    // ---- t = w1 + eB + x3[src] + x4[dst]; store + local BN stats
    int col  = lane & 15;
    int row0 = tile * 16 + ((lane >> 4) << 3);
    float bb0 = eB[col], bb1 = eB[16 + col];
    float sum0 = 0.f, sq0 = 0.f, sum1 = 0.f, sq1 = 0.f;
#pragma unroll
    for (int r2 = 0; r2 < 8; ++r2) {
        int row = row0 + r2;
        int sr = ei[row], dr = ei[NE + row];
        float t0 = c0[r2] + bb0 + x3[(size_t)sr * UNITS + col]      + x4[(size_t)dr * UNITS + col];
        float t1 = c1[r2] + bb1 + x3[(size_t)sr * UNITS + 16 + col] + x4[(size_t)dr * UNITS + 16 + col];
        w1[(size_t)row * UNITS + col]      = t0;
        w1[(size_t)row * UNITS + 16 + col] = t1;
        sum0 += t0; sq0 += t0 * t0;
        sum1 += t1; sq1 += t1 * t1;
    }
    atomicAdd(&s[col], sum0);                 // ds_add_f32
    atomicAdd(&s[UNITS + col], sq0);
    atomicAdd(&s[16 + col], sum1);
    atomicAdd(&s[UNITS + 16 + col], sq1);
    __syncthreads();
    if (threadIdx.x < 2 * UNITS) atomicAdd(&stats[threadIdx.x], s[threadIdx.x]);
}

// --------------------------------------------- BN pass 1 (node): t = x1 + agg/deg
__global__ void node_pre_kernel(float* __restrict__ x1, const float* __restrict__ agg,
                                const float* __restrict__ invdeg, float* __restrict__ stats) {
    __shared__ float s[2 * UNITS];
    if (threadIdx.x < 2 * UNITS) s[threadIdx.x] = 0.0f;
    __syncthreads();
    int idx = blockIdx.x * blockDim.x + threadIdx.x;   // over N*32
    int u = idx & 31;
    float t = x1[idx] + agg[idx] * invdeg[idx >> 5];
    x1[idx] = t;
    atomicAdd(&s[u], t);
    atomicAdd(&s[UNITS + u], t * t);
    __syncthreads();
    if (threadIdx.x < 2 * UNITS) atomicAdd(&stats[threadIdx.x], s[threadIdx.x]);
}

// --------------------------------------------- BN pass 2 + SiLU + residual
__global__ void bn_finish_kernel(const float* __restrict__ t, const float* __restrict__ stats,
                                 const float* __restrict__ g, const float* __restrict__ b,
                                 float* __restrict__ acc, float inv_rows) {
    int idx = blockIdx.x * blockDim.x + threadIdx.x;
    int u = idx & 31;
    float mu  = stats[u] * inv_rows;
    float var = stats[UNITS + u] * inv_rows - mu * mu;
    float bn  = (t[idx] - mu) * rsqrtf(var + BN_EPS) * g[u] + b[u];
    acc[idx] += silu_f(bn);
}

// ---------------------------------------------------------------------------
extern "C" void kernel_launch(void* const* d_in, const int* in_sizes, int n_in,
                              void* d_out, int out_size, void* d_ws, size_t ws_size,
                              hipStream_t stream) {
    const float* x         = (const float*)d_in[0];
    const float* edge_attr = (const float*)d_in[1];
    const float* v_w0      = (const float*)d_in[2];
    const float* v_b0      = (const float*)d_in[3];
    const float* e_w0      = (const float*)d_in[4];
    const float* e_b0      = (const float*)d_in[5];
    const float* vW1 = (const float*)d_in[6];  const float* vB1 = (const float*)d_in[7];
    const float* vW2 = (const float*)d_in[8];  const float* vB2 = (const float*)d_in[9];
    const float* vW3 = (const float*)d_in[10]; const float* vB3 = (const float*)d_in[11];
    const float* vW4 = (const float*)d_in[12]; const float* vB4 = (const float*)d_in[13];
    const float* eW  = (const float*)d_in[14]; const float* eB  = (const float*)d_in[15];
    const float* v_bn_g = (const float*)d_in[16]; const float* v_bn_b = (const float*)d_in[17];
    const float* e_bn_g = (const float*)d_in[18]; const float* e_bn_b = (const float*)d_in[19];
    const int*   edge_index = (const int*)d_in[20];

    float* h = (float*)d_out;                 // h lives in d_out [N,32]

    // workspace layout (all chunks 32B-aligned; agg and stats adjacent so one
    // zero_kernel launch clears both)
    char* wsb = (char*)d_ws;
    size_t off = 0;
    float* w      = (float*)(wsb + off); off += (size_t)NE * UNITS * sizeof(float);
    float* w1     = (float*)(wsb + off); off += (size_t)NE * UNITS * sizeof(float);
    float* xbuf   = (float*)(wsb + off); off += (size_t)4 * NN * UNITS * sizeof(float);
    float* agg    = (float*)(wsb + off); off += (size_t)NN * UNITS * sizeof(float);
    float* stats  = (float*)(wsb + off); off += 4 * UNITS * sizeof(float); // node[0:64], edge[64:128]
    float* invdeg = (float*)(wsb + off); off += (size_t)NN * sizeof(float);
    float* biasP  = (float*)(wsb + off); off += (size_t)DEPTH * 4 * UNITS * sizeof(float);
    v16bf* frag   = (v16bf*)(wsb + off);                                   // DEPTH*5*2*32 fragments

    const int TPB = 256;
    const int gN32 = NN * UNITS / TPB;   // 12500 (exact)
    const int gE32 = NE * UNITS / TPB;   // 200000 (exact)

    // ---- prologue: degrees, weight/bias packing, input projections
    zero_kernel<<<(NN + TPB - 1) / TPB, TPB, 0, stream>>>(invdeg, NN);
    deg_count_kernel<<<NE / TPB, TPB, 0, stream>>>(edge_index, invdeg);
    inv_deg_kernel<<<(NN + TPB - 1) / TPB, TPB, 0, stream>>>(invdeg);
    pack_weights_kernel<<<DEPTH * 5 * 2, 32, 0, stream>>>(vW1, vW2, vW3, vW4, eW, frag);
    pack_bias_kernel<<<DEPTH, 4 * UNITS, 0, stream>>>(vB1, vB2, vB3, vB4, biasP);
    init_h_kernel<<<gN32, TPB, 0, stream>>>(x, v_w0, v_b0, h);
    init_w_kernel<<<gE32, TPB, 0, stream>>>(edge_attr, e_w0, e_b0, w);

    for (int d = 0; d < DEPTH; ++d) {
        const v16bf* fragL = frag + (size_t)(d * 5) * 64;      // layer base
        // clear agg + both stat blocks in one launch (adjacent in ws)
        zero_kernel<<<(NN * UNITS + 4 * UNITS + TPB - 1) / TPB, TPB, 0, stream>>>(
            agg, NN * UNITS + 4 * UNITS);

        node_gemm_kernel<<<(NN / 16) * 4 / 8, TPB, 0, stream>>>(
            h, fragL, biasP + (size_t)d * 4 * UNITS, xbuf);

        edge_fused_kernel<<<(NE / 16) / 8, TPB, 0, stream>>>(
            w, fragL + 4 * 64, eB + d * UNITS,
            xbuf + (size_t)1 * NN * UNITS /*x2*/,
            xbuf + (size_t)2 * NN * UNITS /*x3*/,
            xbuf + (size_t)3 * NN * UNITS /*x4*/,
            edge_index, agg, w1, stats + 2 * UNITS);

        node_pre_kernel<<<gN32, TPB, 0, stream>>>(xbuf /*x1->t*/, agg, invdeg, stats);

        bn_finish_kernel<<<gN32, TPB, 0, stream>>>(
            xbuf, stats, v_bn_g + d * UNITS, v_bn_b + d * UNITS, h, 1.0f / (float)NN);
        bn_finish_kernel<<<gE32, TPB, 0, stream>>>(
            w1, stats + 2 * UNITS, e_bn_g + d * UNITS, e_bn_b + d * UNITS, w, 1.0f / (float)NE);
    }
}
